// TRMamba_89129161326950
// MI455X (gfx1250) — matrix operand order repaired and checked by reading, not compiled, not run on hardware
//
#include <hip/hip_runtime.h>
#include <hip/hip_bf16.h>
#include <math.h>
#include <stdint.h>

typedef __attribute__((ext_vector_type(16))) __bf16 v16bf;
typedef __attribute__((ext_vector_type(8)))  float  v8f;

// exact pointee type expected by the async-copy builtin: int __vector(4)
typedef int v4i_t __attribute__((vector_size(16)));
typedef __attribute__((address_space(1))) v4i_t* gas_v4i;
typedef __attribute__((address_space(3))) v4i_t* las_v4i;

namespace {
constexpr int B    = 512;
constexpr int L    = 64;
constexpr int H    = 128;
constexpr int T    = 128;
constexpr int NENT = 200000;
constexpr int KQ   = 2 * H + T;   // 384: [head|rel|tfeat]
constexpr int KH   = 2 * H;       // 256: [ne|re] (delta part folded analytically)
constexpr int LC   = 16;          // history L-chunk staged in LDS
}

// ---------------------------------------------------------------------------
// CDNA5 async global->LDS copy (ASYNCcnt-tracked), with inline-asm fallback.
// ---------------------------------------------------------------------------
__device__ __forceinline__ void async_copy_b128(const float* gsrc, float* ldst)
{
#if __has_builtin(__builtin_amdgcn_global_load_async_to_lds_b128)
    v4i_t* gp = (v4i_t*)const_cast<float*>(gsrc);
    v4i_t* lp = (v4i_t*)ldst;
    __builtin_amdgcn_global_load_async_to_lds_b128(
        (gas_v4i)gp, (las_v4i)lp, /*offset=*/0, /*cpol=*/0);
#else
    unsigned loff = (unsigned)(uintptr_t)(las_v4i)(v4i_t*)ldst;
    unsigned long long ga = (unsigned long long)(uintptr_t)gsrc;
    asm volatile("global_load_async_to_lds_b128 %0, %1, off"
                 :: "v"(loff), "v"(ga) : "memory");
#endif
}

__device__ __forceinline__ void wait_async0()
{
#if __has_builtin(__builtin_amdgcn_s_wait_asynccnt)
    __builtin_amdgcn_s_wait_asynccnt(0);
#else
    asm volatile("s_wait_asynccnt 0x0" ::: "memory");
#endif
}

// ---------------------------------------------------------------------------
// Phase 1: per-batch encoder. One workgroup per batch element, 128 threads
// (thread h owns feature h). Produces w[b,h] = (q + context) * rel  (fp32).
// ---------------------------------------------------------------------------
__global__ __launch_bounds__(128) void trmamba_encode(
    const int*   __restrict__ heads,
    const int*   __restrict__ relations,
    const int*   __restrict__ times,
    const int*   __restrict__ hist_ent,
    const int*   __restrict__ hist_rel,
    const float* __restrict__ hist_delta,
    const float* __restrict__ hist_mask,
    const float* __restrict__ entity_emb,
    const float* __restrict__ relation_emb,
    const float* __restrict__ time_emb,
    const float* __restrict__ delta_W,
    const float* __restrict__ delta_b,
    const float* __restrict__ query_W,
    const float* __restrict__ query_b,
    const float* __restrict__ in_W,
    const float* __restrict__ in_b,
    const float* __restrict__ a_W,
    const float* __restrict__ a_b,
    const float* __restrict__ hyper_in_W,
    const float* __restrict__ hyper_in_b,
    const float* __restrict__ hyper_gate_W,
    const float* __restrict__ hyper_gate_b,
    const float* __restrict__ attn_W,
    const float* __restrict__ attn_b,
    float*       __restrict__ w_out)
{
    __shared__ float s_cat[KQ];          // [head | rel | tfeat] query input
    __shared__ float s_hist[LC][KH];     // staged [ne|re] rows for LC steps
    __shared__ float s_x[LC][H];         // x rows for current chunk
    __shared__ float s_hseq[L][H + 1];   // gated scan outputs (129 stride: no bank conflict)
    __shared__ float s_scores[L];
    __shared__ float s_attn[L];
    __shared__ float s_red[128];

    const int b = blockIdx.x;
    const int h = threadIdx.x;           // 0..127

    const int rel_i  = relations[b];
    const int head_i = heads[b];
    const int time_i = times[b];

    s_cat[h]         = entity_emb[(size_t)head_i * H + h];
    s_cat[H + h]     = relation_emb[rel_i * H + h];
    s_cat[2 * H + h] = time_emb[time_i * T + h];
    __syncthreads();

    // ---- q = concat(head, rel, tfeat) @ query_W + query_b ----
    float q = query_b[h];
    for (int k = 0; k < KQ; ++k)
        q = fmaf(s_cat[k], query_W[k * H + h], q);

    // ---- hypernet conditioning on rel ----
    float ci = hyper_in_b[h], cg = hyper_gate_b[h];
    for (int k = 0; k < H; ++k) {
        const float r = s_cat[H + k];
        ci = fmaf(r, hyper_in_W[k * H + h], ci);
        cg = fmaf(r, hyper_gate_W[k * H + h], cg);
    }
    cg = 1.f / (1.f + __expf(-cg));

    // ---- fold rank-1 delta feature: df@W3 = hd*u + v per output column ----
    float u = 0.f, v = 0.f;
    for (int t = 0; t < T; ++t) {
        const float wv = in_W[(2 * H + t) * H + h];
        u = fmaf(delta_W[t], wv, u);
        v = fmaf(delta_b[t], wv, v);
    }
    const float xbias = in_b[h] + ci + v;

    // ---- chunked gather -> x GEMM -> a GEMM -> gated scan ----
    float hstate = 0.f;
    for (int c = 0; c < L / LC; ++c) {
        // gather neighbor/relation embedding rows into LDS
        for (int ll = 0; ll < LC; ++ll) {
            const int l = c * LC + ll;
            const int e = hist_ent[b * L + l];
            const int r = hist_rel[b * L + l];
            s_hist[ll][h]     = entity_emb[(size_t)e * H + h];
            s_hist[ll][H + h] = relation_emb[r * H + h];
        }
        __syncthreads();

        // x[ll][h] = hist[ll] @ in_W[:,h] (+bias, cond, delta term)
        float acc[LC];
        #pragma unroll
        for (int ll = 0; ll < LC; ++ll) {
            const float hd = hist_delta[b * L + c * LC + ll];
            acc[ll] = fmaf(hd, u, xbias);
        }
        for (int kc = 0; kc < KH / 16; ++kc) {
            float wr[16];
            #pragma unroll
            for (int kk = 0; kk < 16; ++kk)
                wr[kk] = in_W[(kc * 16 + kk) * H + h];
            #pragma unroll
            for (int ll = 0; ll < LC; ++ll) {
                float s = 0.f;
                #pragma unroll
                for (int kk = 0; kk < 16; ++kk)
                    s = fmaf(s_hist[ll][kc * 16 + kk], wr[kk], s);
                acc[ll] += s;
            }
        }
        #pragma unroll
        for (int ll = 0; ll < LC; ++ll) s_x[ll][h] = acc[ll];
        __syncthreads();

        // a logits: acc[ll] = x[ll] @ a_W[:,h] + a_b[h]
        #pragma unroll
        for (int ll = 0; ll < LC; ++ll) acc[ll] = a_b[h];
        for (int kc = 0; kc < H / 16; ++kc) {
            float wr[16];
            #pragma unroll
            for (int kk = 0; kk < 16; ++kk)
                wr[kk] = a_W[(kc * 16 + kk) * H + h];
            #pragma unroll
            for (int ll = 0; ll < LC; ++ll) {
                float s = 0.f;
                #pragma unroll
                for (int kk = 0; kk < 16; ++kk)
                    s = fmaf(s_x[ll][kc * 16 + kk], wr[kk], s);
                acc[ll] += s;
            }
        }

        // gated recurrence (sequential in l), store gated output
        #pragma unroll
        for (int ll = 0; ll < LC; ++ll) {
            const int l = c * LC + ll;
            const float ah = 1.f / (1.f + __expf(-acc[ll]));
            const float xt = s_x[ll][h];
            const float mt = hist_mask[b * L + l];
            const float hn = ah * hstate + (1.f - ah) * xt;
            hstate = mt * hn + (1.f - mt) * hstate;
            s_hseq[l][h] = hstate * cg;
        }
        __syncthreads();
    }

    // ---- attention pooling ----
    s_red[h] = q * attn_W[H + h];
    __syncthreads();
    for (int st = 64; st > 0; st >>= 1) {
        if (h < st) s_red[h] += s_red[h + st];
        __syncthreads();
    }
    const float qdot = s_red[0];
    __syncthreads();

    const float ab0 = attn_b[0];
    if (h < L) {
        float sc = qdot + ab0;
        for (int k = 0; k < H; ++k)
            sc = fmaf(s_hseq[h][k], attn_W[k], sc);
        sc = tanhf(sc);
        const float mt = hist_mask[b * L + h];
        s_scores[h] = (mt <= 0.f) ? -1e9f : sc;
    }
    __syncthreads();

    s_red[h] = (h < L) ? s_scores[h] : -3.0e38f;
    __syncthreads();
    for (int st = 64; st > 0; st >>= 1) {
        if (h < st) s_red[h] = fmaxf(s_red[h], s_red[h + st]);
        __syncthreads();
    }
    const float mx = s_red[0];
    __syncthreads();

    const float ex = (h < L) ? __expf(s_scores[h] - mx) : 0.f;
    s_red[h] = ex;
    __syncthreads();
    for (int st = 64; st > 0; st >>= 1) {
        if (h < st) s_red[h] += s_red[h + st];
        __syncthreads();
    }
    const float sumex = s_red[0];
    __syncthreads();

    const float p = (h < L) ? (ex / sumex) * hist_mask[b * L + h] : 0.f;
    s_red[h] = p;
    __syncthreads();
    for (int st = 64; st > 0; st >>= 1) {
        if (h < st) s_red[h] += s_red[h + st];
        __syncthreads();
    }
    const float sump = s_red[0];
    __syncthreads();
    if (h < L) s_attn[h] = p / fmaxf(sump, 1e-9f);
    __syncthreads();

    float ctx = 0.f;
    for (int l = 0; l < L; ++l)
        ctx = fmaf(s_hseq[l][h], s_attn[l], ctx);

    w_out[b * H + h] = (q + ctx) * s_cat[H + h];
}

// ---------------------------------------------------------------------------
// Phase 2: out[b,n] = sum_h w[b,h] * E[n,h]  via V_WMMA_F32_16X16X32_BF16.
// The 16-entity B tile (8 KB fp32) is staged into LDS ONCE per workgroup via
// async global->LDS copies (ASYNCcnt), then shared by all 8 waves; A rows are
// loaded direct (w is 256 KB, cache-hot). fp32 accumulate; nan_to_num epilogue.
// ---------------------------------------------------------------------------
__device__ __forceinline__ v16bf pack_bf16(const float4& a, const float4& b,
                                           const float4& c, const float4& d)
{
    v16bf r;
    r[0]  = (__bf16)a.x; r[1]  = (__bf16)a.y; r[2]  = (__bf16)a.z; r[3]  = (__bf16)a.w;
    r[4]  = (__bf16)b.x; r[5]  = (__bf16)b.y; r[6]  = (__bf16)b.z; r[7]  = (__bf16)b.w;
    r[8]  = (__bf16)c.x; r[9]  = (__bf16)c.y; r[10] = (__bf16)c.z; r[11] = (__bf16)c.w;
    r[12] = (__bf16)d.x; r[13] = (__bf16)d.y; r[14] = (__bf16)d.z; r[15] = (__bf16)d.w;
    return r;
}

__global__ __launch_bounds__(256) void distmult_score(
    const float* __restrict__ w,     // [B, H]  fp32 (rep*rel)
    const float* __restrict__ ent,   // [NENT, H] fp32
    float*       __restrict__ out)   // [B, NENT]
{
    constexpr int ROWW = H + 4;          // 132-float row pitch: conflict-free ds_load_b128
    __shared__ float s_b[16 * ROWW];     // B tile: 16 entities x 128 (padded)

    const int t      = threadIdx.x;
    const int lane   = t & 31;
    const int wave   = t >> 5;
    const int lane16 = lane & 15;
    const int hi     = lane >> 4;

    const int n0 = blockIdx.x * 16;               // entity tile
    const int m0 = blockIdx.y * 128 + wave * 16;  // batch tile (8 waves x 16)

    // ---- async stage B tile: 16 rows x 512B = 512 b128 chunks, 2 per thread ----
    {
        const float* gbase = ent + (size_t)n0 * H;
        #pragma unroll
        for (int i = 0; i < 2; ++i) {
            const int f   = t + i * 256;      // chunk id 0..511
            const int row = f >> 5;           // 32 chunks per row
            const int col = (f & 31) << 2;    // float offset within row
            async_copy_b128(gbase + row * H + col, &s_b[row * ROWW + col]);
        }
    }

    const float* Arow = w + (size_t)(m0 + lane16) * H;
    const float* Brow = &s_b[lane16 * ROWW];

    wait_async0();
    __syncthreads();

    v8f c = {0.f, 0.f, 0.f, 0.f, 0.f, 0.f, 0.f, 0.f};

    #pragma unroll
    for (int kc = 0; kc < 4; ++kc) {
        const int kb = kc * 32;
        // A (16x32 bf16): lane half hi -> K runs [kb+8*hi, +8) and [kb+16+8*hi, +8)
        const float4 a0 = *(const float4*)(Arow + kb + 8 * hi);
        const float4 a1 = *(const float4*)(Arow + kb + 8 * hi + 4);
        const float4 a2 = *(const float4*)(Arow + kb + 16 + 8 * hi);
        const float4 a3 = *(const float4*)(Arow + kb + 16 + 8 * hi + 4);
        // B (32x16 bf16) from LDS: lane half hi -> K runs [kb+8*hi, +8) and [kb+16+8*hi, +8)
        const float4 b0 = *(const float4*)(Brow + kb + 8 * hi);
        const float4 b1 = *(const float4*)(Brow + kb + 8 * hi + 4);
        const float4 b2 = *(const float4*)(Brow + kb + 16 + 8 * hi);
        const float4 b3 = *(const float4*)(Brow + kb + 16 + 8 * hi + 4);

        const v16bf av = pack_bf16(a0, a1, a2, a3);
        const v16bf bv = pack_bf16(b0, b1, b2, b3);

        c = __builtin_amdgcn_wmma_f32_16x16x32_bf16(
                /*neg_a=*/false, av, /*neg_b=*/false, bv,
                /*c_mod=*/(short)0, c, /*reuse_a=*/false, /*reuse_b=*/false);
    }

    #pragma unroll
    for (int r = 0; r < 8; ++r) {
        float vr = c[r];
        if (__builtin_isnan(vr))      vr = 0.f;
        else if (__builtin_isinf(vr)) vr = (vr > 0.f) ? 1e6f : -1e6f;
        const size_t m = (size_t)(m0 + r + 8 * hi);
        out[m * (size_t)NENT + (n0 + lane16)] = vr;
    }
}

// ---------------------------------------------------------------------------
extern "C" void kernel_launch(void* const* d_in, const int* in_sizes, int n_in,
                              void* d_out, int out_size, void* d_ws, size_t ws_size,
                              hipStream_t stream)
{
    const int*   heads        = (const int*)  d_in[0];
    const int*   relations    = (const int*)  d_in[1];
    const int*   times        = (const int*)  d_in[2];
    const int*   hist_ent     = (const int*)  d_in[3];
    const int*   hist_rel     = (const int*)  d_in[4];
    const float* hist_delta   = (const float*)d_in[5];
    const float* hist_mask    = (const float*)d_in[6];
    const float* entity_emb   = (const float*)d_in[7];
    const float* relation_emb = (const float*)d_in[8];
    const float* time_emb     = (const float*)d_in[9];
    const float* delta_W      = (const float*)d_in[10];
    const float* delta_b      = (const float*)d_in[11];
    const float* query_W      = (const float*)d_in[12];
    const float* query_b      = (const float*)d_in[13];
    const float* in_W         = (const float*)d_in[14];
    const float* in_b         = (const float*)d_in[15];
    const float* a_W          = (const float*)d_in[16];
    const float* a_b          = (const float*)d_in[17];
    const float* hyper_in_W   = (const float*)d_in[18];
    const float* hyper_in_b   = (const float*)d_in[19];
    const float* hyper_gate_W = (const float*)d_in[20];
    const float* hyper_gate_b = (const float*)d_in[21];
    const float* attn_W       = (const float*)d_in[22];
    const float* attn_b       = (const float*)d_in[23];

    float* w = (float*)d_ws;   // [B, H] = 256 KB scratch

    trmamba_encode<<<dim3(B), dim3(H), 0, stream>>>(
        heads, relations, times, hist_ent, hist_rel, hist_delta, hist_mask,
        entity_emb, relation_emb, time_emb, delta_W, delta_b,
        query_W, query_b, in_W, in_b, a_W, a_b,
        hyper_in_W, hyper_in_b, hyper_gate_W, hyper_gate_b,
        attn_W, attn_b, w);

    distmult_score<<<dim3(NENT / 16, B / 128), dim3(256), 0, stream>>>(
        w, entity_emb, (float*)d_out);
}
